// MultiHeadCrossAttention_70686571758276
// MI455X (gfx1250) — matrix operand order repaired and checked by reading, not compiled, not run on hardware
//
#include <hip/hip_runtime.h>
#include <cstdint>

// ---------------------------------------------------------------------------
// Problem constants (from reference): B=4, S=2048, H=1024, 4 heads, d=256
// ---------------------------------------------------------------------------
#define B_   4
#define S_   2048
#define H_   1024
#define NH_  4
#define D_   256
#define BH_  (B_ * NH_)

typedef __attribute__((ext_vector_type(16))) _Float16 v16h;
typedef __attribute__((ext_vector_type(8)))  _Float16 v8h;
typedef __attribute__((ext_vector_type(4)))  _Float16 v4h;
typedef __attribute__((ext_vector_type(8)))  float    v8f;
typedef __attribute__((ext_vector_type(4)))  float    v4f;
typedef __attribute__((ext_vector_type(4)))  unsigned int u32x4;
typedef __attribute__((ext_vector_type(8)))  int          i32x8;
typedef __attribute__((ext_vector_type(4)))  int          i32x4;

union F16x16 { v16h v; v8h h[2]; };

__device__ __forceinline__ v8f zero8() {
  v8f v;
#pragma unroll
  for (int i = 0; i < 8; ++i) v[i] = 0.0f;
  return v;
}

// ---------------------------------------------------------------------------
// TDM 2D tile load: build D# (ISA 8.3/8.4) and issue TENSOR_LOAD_TO_LDS.
//   group0: [1:0]=count=1 | [63:32]=lds byte addr | [120:64]=global addr |
//           [127:126]=type=2
//   group1: [17:16]=data_size(1 => 2B) | [79:48]=tensor_dim0 |
//           [111:80]=tensor_dim1 | [127:112]=tile_dim0 | [143:128]=tile_dim1 |
//           [207:160]=tensor_dim0_stride  (strides/dims in elements)
// lds_addr = low 32 bits of flat shared pointer (flat LDS aperture keeps the
// LDS byte offset in addr[31:0], ISA 10.2). Tracked by TENSORcnt.
// This toolchain exposes the 6-arg builtin:
//   (u32x4 g0, i32x8 g1, i32x4 g2, i32x4 g3, i32x8 extra, i32 cpol)
// groups 2/3/extra are zero for 2D tiles.
// ---------------------------------------------------------------------------
__device__ __forceinline__ void tdm_load_2d(
    const _Float16* lds_dst, const _Float16* gsrc,
    unsigned tensor_d0, unsigned tensor_d1,
    unsigned tile_d0, unsigned tile_d1, unsigned stride0)
{
  unsigned long long ga = (unsigned long long)(uintptr_t)gsrc;
  unsigned lds_off = (unsigned)(uintptr_t)lds_dst;
  u32x4 g0;
  g0[0] = 1u;                                           // count=1, user D#
  g0[1] = lds_off;
  g0[2] = (unsigned)(ga & 0xFFFFFFFFu);
  g0[3] = (unsigned)((ga >> 32) & 0x01FFFFFFu) | 0x80000000u;  // type=2
  i32x8 g1;
  g1[0] = 0x00010000;                                   // data_size=1 (2B)
  g1[1] = (int)((tensor_d0 & 0xFFFFu) << 16);
  g1[2] = (int)(((tensor_d0 >> 16) & 0xFFFFu) | ((tensor_d1 & 0xFFFFu) << 16));
  g1[3] = (int)(((tensor_d1 >> 16) & 0xFFFFu) | ((tile_d0 & 0xFFFFu) << 16));
  g1[4] = (int)(tile_d1 & 0xFFFFu);                     // tile_dim2 = 0
  g1[5] = (int)stride0;                                 // dim0 stride low32
  g1[6] = 0;                                            // stride hi / dim1 stride
  g1[7] = 0;
  i32x4 z4;
  z4[0] = 0; z4[1] = 0; z4[2] = 0; z4[3] = 0;
  i32x8 z8;
#pragma unroll
  for (int i = 0; i < 8; ++i) z8[i] = 0;
  __builtin_amdgcn_tensor_load_to_lds(g0, g1, z4, z4, z8, 0);
}

// ---------------------------------------------------------------------------
// Kernel 1: FiLM modulation, head split, f32 -> f16 pack.
//   qh[bh][s][dd], kh[bh][s][dd] row-major; vh[bh][dd][s] TRANSPOSED so the
//   flash kernel can TDM-load V tiles d-major (no in-kernel transpose).
// ---------------------------------------------------------------------------
__global__ __launch_bounds__(256) void qkv_mod_kernel(
    const float* __restrict__ relation, const float* __restrict__ samples,
    const float* __restrict__ W, _Float16* __restrict__ qh,
    _Float16* __restrict__ kh, _Float16* __restrict__ vh)
{
  int idx = blockIdx.x * 256 + threadIdx.x;   // one float4 per thread
  int c4  = idx * 4;
  int b   = c4 / (S_ * H_);
  int rem = c4 - b * (S_ * H_);
  int s   = rem / H_;
  int c   = rem - s * H_;
  int head = c / D_;
  int dd   = c - head * D_;

  v4f sm = *(const v4f*)(samples + (size_t)c4);
  v4f rl = *(const v4f*)(relation + (size_t)c4);
  const float* wb = W + (size_t)b * 8 * H_ + c;
  v4f w0 = *(const v4f*)(wb + 0 * H_);
  v4f w1 = *(const v4f*)(wb + 1 * H_);
  v4f w2 = *(const v4f*)(wb + 2 * H_);
  v4f w3 = *(const v4f*)(wb + 3 * H_);
  v4f w4 = *(const v4f*)(wb + 4 * H_);
  v4f w5 = *(const v4f*)(wb + 5 * H_);

  v4h q, k;
  _Float16 v[4];
#pragma unroll
  for (int j = 0; j < 4; ++j) {
    q[j] = (_Float16)(sm[j] * w0[j] + w1[j]);
    k[j] = (_Float16)(rl[j] * w2[j] + w3[j]);
    v[j] = (_Float16)(rl[j] * w4[j] + w5[j]);
  }
  int bh = b * NH_ + head;
  size_t o = ((size_t)bh * S_ + s) * D_ + dd;
  *(v4h*)(qh + o) = q;
  *(v4h*)(kh + o) = k;
  size_t vo = ((size_t)bh * D_ + dd) * S_ + s;   // transposed [bh][d][s]
#pragma unroll
  for (int j = 0; j < 4; ++j) vh[vo + (size_t)j * S_] = v[j];
}

// ---------------------------------------------------------------------------
// Kernel 2: flash attention (no 1/sqrt(d) scale, matching reference).
// Grid: (S/64, B*NH). 128 threads = 4 waves; each wave owns 16 q rows, full d.
// KV staged 32 rows/stage by the Tensor Data Mover, double-buffered:
//   Kt[buf]: row-major [kv][d]   (TDM tile 256x32, stride 256)
//   Vt[buf]: d-major  [d][kv]    (TDM tile 32x256, stride 2048 from vh^T)
// WMMA fragment layouts per CDNA5 ISA 7.12.2 (wave32):
//   A 16x32 f16 : lane l = row l%16; halves = K runs [a,a+8),[a+16,a+24),
//                 a=(l/16)*8
//   B 32x16 f16 : lane l = col l%16; halves = K runs [a,a+8),[a+16,a+24), a=(l/16)*8
//   C/D 16x16 f32: VGPR r, lane l -> (row r+8*(l/16), col l%16)
// ---------------------------------------------------------------------------
#define KVT 32

__global__ __launch_bounds__(128) void flash_attn_kernel(
    const _Float16* __restrict__ qh, const _Float16* __restrict__ kh,
    const _Float16* __restrict__ vh, float* __restrict__ ctx)
{
  __shared__ _Float16 Kt[2][KVT * D_];     // 2 x 16 KB
  __shared__ _Float16 Vt[2][D_ * KVT];     // 2 x 16 KB
  __shared__ _Float16 Pl[4][16 * 32];      // 4 KB per-wave P staging

  const int tid  = threadIdx.x;
  const int w    = tid >> 5;
  const int lane = tid & 31;
  const int l16  = lane & 15;
  const int lhi  = lane >> 4;
  const int bh   = blockIdx.y;
  const int b    = bh >> 2;
  const int head = bh & 3;
  const int qrow0 = blockIdx.x * 64 + w * 16;

  // --- load Q A-fragments (16 rows x 256 d) into registers: 8 x v16h -------
  v16h Aq[8];
  {
    const _Float16* qrow = qh + ((size_t)bh * S_ + qrow0 + l16) * D_;
#pragma unroll
    for (int ch = 0; ch < 8; ++ch) {
      F16x16 u;
      int a0 = ch * 32 + lhi * 8;
      u.h[0] = *(const v8h*)(qrow + a0);
      u.h[1] = *(const v8h*)(qrow + a0 + 16);
      Aq[ch] = u.v;
    }
  }

  v8f Od[16];                      // 16 rows x 256 d f32 accumulator
#pragma unroll
  for (int i = 0; i < 16; ++i) Od[i] = zero8();
  float mrow[8], lrow[8];
#pragma unroll
  for (int r = 0; r < 8; ++r) { mrow[r] = -3.0e38f; lrow[r] = 0.0f; }

  const _Float16* kbase = kh + (size_t)bh * S_ * D_;   // [kv][d]
  const _Float16* vbase = vh + (size_t)bh * D_ * S_;   // [d][kv]

  const int NST = S_ / KVT;                            // 64 stages

  // prologue: TDM stage 0 into buffer 0 (one wave issues; EXEC-independent op)
  if (w == 0) {
    tdm_load_2d(&Kt[0][0], kbase, D_, S_, D_, KVT, D_);
    tdm_load_2d(&Vt[0][0], vbase, S_, D_, KVT, D_, S_);
  }
  __builtin_amdgcn_s_wait_tensorcnt(0);
  __syncthreads();

  for (int st = 0; st < NST; ++st) {
    const int buf = st & 1;
    // kick off next stage's DMA into the other buffer (overlaps the WMMAs)
    if (w == 0 && st + 1 < NST) {
      int kv1 = (st + 1) * KVT;
      tdm_load_2d(&Kt[buf ^ 1][0], kbase + (size_t)kv1 * D_, D_, S_, D_, KVT, D_);
      tdm_load_2d(&Vt[buf ^ 1][0], vbase + kv1,              S_, D_, KVT, D_, S_);
    }

    // ---- S = Q K^T for two adjacent 16-col tiles (32 kv columns) ----------
    v8f sa[2];
#pragma unroll
    for (int sub = 0; sub < 2; ++sub) {
      int n16 = sub * 16;
      v8f acc = zero8();
#pragma unroll
      for (int ch = 0; ch < 8; ++ch) {
        F16x16 ub;
        const _Float16* krow = &Kt[buf][(n16 + l16) * D_ + ch * 32 + lhi * 8];
        ub.h[0] = *(const v8h*)(krow);
        ub.h[1] = *(const v8h*)(krow + 16);
        acc = __builtin_amdgcn_wmma_f32_16x16x32_f16(
                  false, Aq[ch], false, ub.v, (short)0, acc, false, false);
      }
      sa[sub] = acc;
    }

    // ---- online softmax over the 32 new columns ---------------------------
    float av[8];
#pragma unroll
    for (int r = 0; r < 8; ++r) {
      float mx = fmaxf(sa[0][r], sa[1][r]);
      mx = fmaxf(mx, __shfl_xor(mx, 1));
      mx = fmaxf(mx, __shfl_xor(mx, 2));
      mx = fmaxf(mx, __shfl_xor(mx, 4));
      mx = fmaxf(mx, __shfl_xor(mx, 8));
      float mn    = fmaxf(mrow[r], mx);
      float alpha = __expf(mrow[r] - mn);
      float p0    = __expf(sa[0][r] - mn);
      float p1    = __expf(sa[1][r] - mn);
      float ps = p0 + p1;
      ps += __shfl_xor(ps, 1);
      ps += __shfl_xor(ps, 2);
      ps += __shfl_xor(ps, 4);
      ps += __shfl_xor(ps, 8);
      lrow[r] = lrow[r] * alpha + ps;
      mrow[r] = mn;
      av[r]   = alpha;
      int prow = (r + 8 * lhi) * 32;           // D-layout row -> P row
      Pl[w][prow + l16]      = (_Float16)p0;
      Pl[w][prow + 16 + l16] = (_Float16)p1;
    }
#pragma unroll
    for (int dc = 0; dc < 16; ++dc)
#pragma unroll
      for (int r = 0; r < 8; ++r) Od[dc][r] *= av[r];

    // ---- O += P V  (A = P 16x32 from LDS, B = V 32x16 from Vt d-major) ----
#pragma unroll
    for (int dc = 0; dc < 16; ++dc) {
      F16x16 ua, ub;
      const _Float16* prow = &Pl[w][l16 * 32 + lhi * 8];
      ua.h[0] = *(const v8h*)(prow);
      ua.h[1] = *(const v8h*)(prow + 16);
      const _Float16* vrow = &Vt[buf][(dc * 16 + l16) * KVT + lhi * 8];
      ub.h[0] = *(const v8h*)(vrow);
      ub.h[1] = *(const v8h*)(vrow + 16);
      Od[dc] = __builtin_amdgcn_wmma_f32_16x16x32_f16(
                   false, ua.v, false, ub.v, (short)0, Od[dc], false, false);
    }

    // close pipeline stage: DMA done (issuing wave), LDS visible to all waves
    __builtin_amdgcn_s_wait_tensorcnt(0);
    __syncthreads();
  }

  // ---- normalize and store ctx in merged-head [B][Sq][H] layout -----------
  float rinv[8];
#pragma unroll
  for (int r = 0; r < 8; ++r) rinv[r] = 1.0f / lrow[r];
#pragma unroll
  for (int dc = 0; dc < 16; ++dc) {
#pragma unroll
    for (int r = 0; r < 8; ++r) {
      int m   = qrow0 + r + 8 * lhi;
      int col = head * D_ + dc * 16 + l16;
      ctx[((size_t)b * S_ + m) * H_ + col] = Od[dc][r] * rinv[r];
    }
  }
}

// ---------------------------------------------------------------------------
// Kernel 3: x = gelu(ctx) + samples; LayerNorm(H) no-affine; *W6 + W7.
// ---------------------------------------------------------------------------
__global__ __launch_bounds__(256) void epilogue_kernel(
    const float* __restrict__ ctx, const float* __restrict__ samples,
    const float* __restrict__ W, float* __restrict__ out)
{
  int bs = blockIdx.x;
  int b  = bs >> 11;                     // / 2048
  const float* crow = ctx + (size_t)bs * H_;
  const float* srow = samples + (size_t)bs * H_;
  int t = threadIdx.x;

  float x[4];
  float sum = 0.0f, sq = 0.0f;
#pragma unroll
  for (int i = 0; i < 4; ++i) {
    int c = t + i * 256;
    float g  = crow[c];
    float u  = 0.7978845608028654f * (g + 0.044715f * g * g * g);
    float ge = 0.5f * g * (1.0f + tanhf(u));   // jax.nn.gelu (tanh approx)
    float xv = ge + srow[c];
    x[i] = xv; sum += xv; sq += xv * xv;
  }
#pragma unroll
  for (int mask = 1; mask < 32; mask <<= 1) {
    sum += __shfl_xor(sum, mask);
    sq  += __shfl_xor(sq, mask);
  }
  __shared__ float rs[8], rq[8];
  int w = t >> 5, lane = t & 31;
  if (lane == 0) { rs[w] = sum; rq[w] = sq; }
  __syncthreads();
  float ts = 0.0f, tq = 0.0f;
#pragma unroll
  for (int i = 0; i < 8; ++i) { ts += rs[i]; tq += rq[i]; }
  float mu   = ts * (1.0f / H_);
  float var  = tq * (1.0f / H_) - mu * mu;
  float rstd = rsqrtf(var + 1e-12f);

  const float* w6 = W + ((size_t)b * 8 + 6) * H_;
  const float* w7 = W + ((size_t)b * 8 + 7) * H_;
  float* orow = out + (size_t)bs * H_;
#pragma unroll
  for (int i = 0; i < 4; ++i) {
    int c = t + i * 256;
    orow[c] = (x[i] - mu) * rstd * w6[c] + w7[c];
  }
}

// ---------------------------------------------------------------------------
extern "C" void kernel_launch(void* const* d_in, const int* in_sizes, int n_in,
                              void* d_out, int out_size, void* d_ws, size_t ws_size,
                              hipStream_t stream)
{
  const float* relation = (const float*)d_in[0];
  const float* samples  = (const float*)d_in[1];
  const float* W        = (const float*)d_in[2];
  float* out = (float*)d_out;

  // workspace: qh | kh (row-major f16) | vh (d-major f16) | ctx (f32) ~84 MB
  size_t nQ = (size_t)B_ * NH_ * S_ * D_;
  _Float16* qh = (_Float16*)d_ws;
  _Float16* kh = qh + nQ;
  _Float16* vh = kh + nQ;
  float*    ctx = (float*)(vh + nQ);

  qkv_mod_kernel<<<(B_ * S_ * H_ / 4) / 256, 256, 0, stream>>>(
      relation, samples, W, qh, kh, vh);
  flash_attn_kernel<<<dim3(S_ / 64, BH_), 128, 0, stream>>>(qh, kh, vh, ctx);
  epilogue_kernel<<<B_ * S_, 256, 0, stream>>>(ctx, samples, W, out);
}
